// CosHead_63668595196214
// MI455X (gfx1250) — matrix operand order repaired and checked by reading, not compiled, not run on hardware
//
#include <hip/hip_runtime.h>

// CosHead on MI455X (gfx1250):
//   out[c,p] = (w[c] . x[:,p]) * xinv[p] * winv[c]
//   xinv[p] = 1/max(||x[:,p]||, 1e-8), winv[c] = scale[c]*5/max(||w[c]||, 1e-8)
// GEMM 256x256x65536 in f16 via v_wmma_f32_16x16x32_f16, fp32 accumulate.
// Memory-bound: 64MB x read + 64MB out write ~ 5.5us floor @ 23.3 TB/s.

#define DLAT 256
#define CNUM 256
#define HW   65536
#define EPSV 1e-8f

typedef _Float16 v8h  __attribute__((ext_vector_type(8)));
typedef _Float16 v16h __attribute__((ext_vector_type(16)));
typedef _Float16 h4   __attribute__((ext_vector_type(4)));
typedef float    v8f  __attribute__((ext_vector_type(8)));

// ---------------- pass 1: norms ----------------

__global__ __launch_bounds__(256) void coshead_xnorm(const float* __restrict__ x,
                                                     float* __restrict__ xinv) {
    const int p = blockIdx.x * 256 + threadIdx.x;   // coalesced over p
    float s = 0.f;
#pragma unroll 8
    for (int d = 0; d < DLAT; ++d) {
        const float v = x[d * HW + p];
        s = fmaf(v, v, s);
    }
    xinv[p] = 1.f / fmaxf(sqrtf(s), EPSV);
}

__global__ __launch_bounds__(256) void coshead_wnorm(const float* __restrict__ w,
                                                     const float* __restrict__ scale,
                                                     float* __restrict__ winv) {
    const int c = threadIdx.x;
    float s = 0.f;
#pragma unroll 8
    for (int d = 0; d < DLAT; ++d) {
        const float v = w[c * DLAT + d];
        s = fmaf(v, v, s);
    }
    winv[c] = scale[c] * 5.0f / fmaxf(sqrtf(s), EPSV);
}

// ---------------- pass 2: WMMA GEMM ----------------
// LDS chunk layout for both operands: buf[q][row][8] halves, q = k>>3.
//   A frag (16x32 f16, ISA layout): lane: row = c&15, chunks q = kt*4+(lane>>4), +2
//   B frag (32x16 f16):             lane: row = p&15 + tile, chunks q = kt*4+(lane>>4), +2

__device__ __forceinline__ v16h load_frag(const _Float16* base, int row, int q1, int q2) {
    const v8h lo = *(const v8h*)(base + (q1 * 128 + row) * 8);   // ds_load_b128
    const v8h hi = *(const v8h*)(base + (q2 * 128 + row) * 8);   // ds_load_b128
    return __builtin_shufflevector(lo, hi, 0,1,2,3,4,5,6,7,8,9,10,11,12,13,14,15);
}

__global__ __launch_bounds__(512) void coshead_gemm(const float* __restrict__ x,
                                                    const float* __restrict__ w,
                                                    const float* __restrict__ xinv,
                                                    const float* __restrict__ winv,
                                                    float* __restrict__ out) {
    __shared__ __align__(16) _Float16 wlds[32 * 128 * 8];   // 64 KB: 128 C rows, K=256
    __shared__ __align__(16) _Float16 xlds[32 * 128 * 8];   // 64 KB: 128 pixels,  K=256

    const int t  = threadIdx.x;
    const int p0 = blockIdx.x * 128;
    const int c0 = blockIdx.y * 128;

    // Pack weights -> f16 LDS chunks (coalesced-in-k float4 reads, 8B LDS stores)
    {
        const int ci = t & 7;
        const int k0 = (t >> 3) * 4;            // 0..252
#pragma unroll
        for (int pass = 0; pass < 16; ++pass) {
            const int c = ci + pass * 8;        // 0..127
            const float4 v = *(const float4*)(w + (c0 + c) * DLAT + k0);
            h4 h = { (_Float16)v.x, (_Float16)v.y, (_Float16)v.z, (_Float16)v.w };
            *(h4*)&wlds[((k0 >> 3) * 128 + c) * 8 + (k0 & 7)] = h;
        }
    }
    // Pack x tile -> f16 LDS chunks (coalesced-in-p b32 reads, register transpose)
    {
        const int p  = t & 127;
        const int kb = (t >> 7) * 4;            // 0,4,8,12
#pragma unroll
        for (int pass = 0; pass < 16; ++pass) {
            const int k = pass * 16 + kb;
            const float a0 = x[(k + 0) * HW + p0 + p];
            const float a1 = x[(k + 1) * HW + p0 + p];
            const float a2 = x[(k + 2) * HW + p0 + p];
            const float a3 = x[(k + 3) * HW + p0 + p];
            h4 h = { (_Float16)a0, (_Float16)a1, (_Float16)a2, (_Float16)a3 };
            *(h4*)&xlds[((k >> 3) * 128 + p) * 8 + (k & 7)] = h;
        }
    }
    __syncthreads();

    const int wave = t >> 5;
    const int lane = t & 31;
    const int lhi  = lane >> 4;
    const int l15  = lane & 15;
    const int pt   = wave & 7;        // 8 pixel tiles of 16
    const int cg   = wave >> 3;       // 2 groups of 4 C tiles
    const int prow = pt * 16 + l15;   // B-fragment LDS row

    const v8f zero = {0.f, 0.f, 0.f, 0.f, 0.f, 0.f, 0.f, 0.f};
    v8f acc[4] = {zero, zero, zero, zero};

#pragma unroll
    for (int kt = 0; kt < 8; ++kt) {                       // K = 256 in steps of 32
        const int bq = kt * 4 + lhi;
        const v16h B = load_frag(xlds, prow, bq, bq + 2);
        const int aq = kt * 4 + lhi;
#pragma unroll
        for (int ci = 0; ci < 4; ++ci) {
            const int arow = (cg * 4 + ci) * 16 + l15;
            const v16h A = load_frag(wlds, arow, aq, aq + 2);
            acc[ci] = __builtin_amdgcn_wmma_f32_16x16x32_f16(
                false, A, false, B, (short)0, acc[ci], false, false);
        }
    }

    // Epilogue: fold cosine normalization + scale*RANGE_EXTENDER, coalesced stores.
    const float xs = xinv[p0 + prow];
    float* outp = out + p0 + prow;
#pragma unroll
    for (int ci = 0; ci < 4; ++ci) {
        const int cbase = c0 + (cg * 4 + ci) * 16 + lhi * 8;
#pragma unroll
        for (int r = 0; r < 8; ++r) {                      // C/D layout: VGPR r -> M=r(+8)
            const int c = cbase + r;
            outp[c * HW] = acc[ci][r] * xs * winv[c];
        }
    }
}

extern "C" void kernel_launch(void* const* d_in, const int* in_sizes, int n_in,
                              void* d_out, int out_size, void* d_ws, size_t ws_size,
                              hipStream_t stream) {
    (void)in_sizes; (void)n_in; (void)out_size; (void)ws_size;
    const float* x     = (const float*)d_in[0];   // (256, 256, 256)
    const float* wts   = (const float*)d_in[1];   // (256, 256)
    const float* scale = (const float*)d_in[2];   // (256,)
    float* out  = (float*)d_out;                  // (256, 65536)
    float* xinv = (float*)d_ws;                   // 65536 floats
    float* winv = xinv + HW;                      // 256 floats

    coshead_xnorm<<<HW / 256, 256, 0, stream>>>(x, xinv);
    coshead_wnorm<<<1, 256, 0, stream>>>(wts, scale, winv);
    coshead_gemm<<<dim3(HW / 128, CNUM / 128), 512, 0, stream>>>(x, wts, xinv, winv, out);
}
